// TinyHFFP8Ref_2731599200928
// MI455X (gfx1250) — 8-bit matrix operand order repaired and checked by reading, compiled, not run on hardware
//
#include <hip/hip_runtime.h>

typedef int   v16i __attribute__((ext_vector_type(16)));
typedef float v8f  __attribute__((ext_vector_type(8)));

#define QBLK     128
#define FP8MAX   448.0f
#define F32TINY  1.17549435e-38f

#if defined(__has_builtin)
#if __has_builtin(__builtin_amdgcn_global_load_async_to_lds_b128)
#define HAVE_ASYNC_LDS 1
#endif
#endif

#if defined(HAVE_ASYNC_LDS)
typedef int v4i_b __attribute__((vector_size(4 * sizeof(int))));
typedef __attribute__((address_space(1))) v4i_b glob_b128_t;
typedef __attribute__((address_space(3))) v4i_b lds_b128_t;

__device__ __forceinline__ void async_copy16(const unsigned char* gsrc, void* ldst) {
    glob_b128_t* g = (glob_b128_t*)(unsigned long long)(const void*)gsrc;
    lds_b128_t*  l = (lds_b128_t*)(unsigned int)(unsigned long long)ldst;
    __builtin_amdgcn_global_load_async_to_lds_b128(g, l, 0, 0);
}

__device__ __forceinline__ void wait_async0() {
#if __has_builtin(__builtin_amdgcn_s_wait_asynccnt)
    __builtin_amdgcn_s_wait_asynccnt(0);
#else
    asm volatile("s_wait_asynccnt 0x0" ::: "memory");
#endif
}
#endif

__device__ __forceinline__ unsigned char f32_to_e4m3_sw(float f) {
    unsigned ub   = __float_as_uint(f);
    unsigned sign = (ub >> 31) << 7;
    float a = fabsf(f);
    if (!(a > 0.0f)) return (unsigned char)sign;
    a = fminf(a, FP8MAX);
    int e; float mfr = frexpf(a, &e);
    int e2 = e - 1;
    unsigned byte;
    if (e2 < -6) {
        int q = (int)(a * 512.0f + 0.5f);
        if (q > 7) q = 7;
        byte = (unsigned)q;
    } else {
        int mant = (int)((mfr * 2.0f - 1.0f) * 8.0f + 0.5f);
        if (mant == 8) { mant = 0; e2++; }
        if (e2 > 8) { e2 = 8; mant = 7; }
        byte = ((unsigned)(e2 + 7) << 3) | (unsigned)mant;
    }
    return (unsigned char)(sign | byte);
}

__device__ __forceinline__ unsigned pack4_fp8(float a, float b, float c, float d) {
#if defined(__has_builtin) && __has_builtin(__builtin_amdgcn_cvt_pk_fp8_f32)
    int v = 0;
    v = __builtin_amdgcn_cvt_pk_fp8_f32(a, b, v, false);
    v = __builtin_amdgcn_cvt_pk_fp8_f32(c, d, v, true);
    return (unsigned)v;
#else
    return (unsigned)f32_to_e4m3_sw(a)
         | ((unsigned)f32_to_e4m3_sw(b) << 8)
         | ((unsigned)f32_to_e4m3_sw(c) << 16)
         | ((unsigned)f32_to_e4m3_sw(d) << 24);
#endif
}

__global__ __launch_bounds__(256)
void quant_act_kernel(const float* __restrict__ x, unsigned char* __restrict__ xq,
                      float* __restrict__ xs, int Kdim, int kbc) {
    const int g    = blockIdx.x * 8 + (threadIdx.x >> 5);
    const int lane = threadIdx.x & 31;
    const int row  = g / kbc;
    const int kb   = g % kbc;
    const size_t base = (size_t)row * Kdim + (size_t)kb * QBLK + lane * 4;
    const float4 v = *(const float4*)(x + base);
    float m = fmaxf(fmaxf(fabsf(v.x), fabsf(v.y)), fmaxf(fabsf(v.z), fabsf(v.w)));
#pragma unroll
    for (int off = 16; off > 0; off >>= 1)
        m = fmaxf(m, __shfl_xor(m, off, 32));
    const float scale = fmaxf(m * (1.0f / FP8MAX), F32TINY);
    const float r = 1.0f / scale;
    *(unsigned*)(xq + base) = pack4_fp8(v.x * r, v.y * r, v.z * r, v.w * r);
    if (lane == 0) xs[(size_t)row * kbc + kb] = scale;
}

__global__ __launch_bounds__(256)
void quant_wgt_kernel(const float* __restrict__ w, unsigned char* __restrict__ wq,
                      float* __restrict__ ws, int Kdim, int kbc) {
    const int nb  = blockIdx.x / kbc;
    const int kb  = blockIdx.x % kbc;
    const int tid = threadIdx.x;
    const int row = nb * QBLK + (tid >> 1);
    const int col = kb * QBLK + (tid & 1) * 64;
    const float* base = w + (size_t)row * Kdim + col;

    float4 vals[16];
    float m = 0.0f;
#pragma unroll
    for (int i = 0; i < 16; ++i) {
        vals[i] = ((const float4*)base)[i];
        m = fmaxf(m, fmaxf(fmaxf(fabsf(vals[i].x), fabsf(vals[i].y)),
                           fmaxf(fabsf(vals[i].z), fabsf(vals[i].w))));
    }
    __shared__ float red[256];
    red[tid] = m;
    __syncthreads();
#pragma unroll
    for (int s = 128; s > 0; s >>= 1) {
        if (tid < s) red[tid] = fmaxf(red[tid], red[tid + s]);
        __syncthreads();
    }
    const float scale = fmaxf(red[0] * (1.0f / FP8MAX), F32TINY);
    const float r = 1.0f / scale;
    unsigned* qb = (unsigned*)(wq + (size_t)row * Kdim + col);
#pragma unroll
    for (int i = 0; i < 16; ++i)
        qb[i] = pack4_fp8(vals[i].x * r, vals[i].y * r, vals[i].z * r, vals[i].w * r);
    if (tid == 0) ws[(size_t)nb * kbc + kb] = scale;
}

__global__ __launch_bounds__(512)
void gemm_fp8_kernel(const unsigned char* __restrict__ Afp8,
                     const unsigned char* __restrict__ Bfp8,
                     const float* __restrict__ Ascale,
                     const float* __restrict__ Bscale,
                     const float* __restrict__ bias,
                     float* __restrict__ out,
                     int Ndim, int Kdim, int kbc) {
    __shared__ __align__(16) unsigned char As[2][QBLK * QBLK];
    __shared__ __align__(16) unsigned char Bs[2][QBLK * QBLK];

    const int tid  = threadIdx.x;
    const int lane = tid & 31;
    const int wave = tid >> 5;
    const int wsm  = wave & 7;
    const int wsn  = wave >> 3;
    const int lhi  = lane >> 4;
    const int llo  = lane & 15;

    const int m0 = blockIdx.x * QBLK;
    const int n0 = blockIdx.y * QBLK;
    const int nb = n0 >> 7;

    const int srow = tid >> 3;
    const int scol = (tid & 7) * 16;

    const unsigned char* Abase = Afp8 + (size_t)m0 * Kdim;
    const unsigned char* Bbase = Bfp8 + (size_t)n0 * Kdim;

    float acc[4][8];
#pragma unroll
    for (int g = 0; g < 4; ++g)
#pragma unroll
        for (int i = 0; i < 8; ++i) acc[g][i] = 0.0f;

#if !defined(HAVE_ASYNC_LDS)
    uint4 ra[2], rb[2];
#endif

#if defined(HAVE_ASYNC_LDS)
#define STAGE_ISSUE(BUF, KB)                                                     \
    {                                                                            \
        _Pragma("unroll")                                                        \
        for (int r = 0; r < 2; ++r) {                                            \
            const int row = srow + 64 * r;                                       \
            const size_t ko = (size_t)(KB) * QBLK + scol;                        \
            async_copy16(Abase + (size_t)row * Kdim + ko,                        \
                         &As[BUF][row * QBLK + scol]);                           \
            async_copy16(Bbase + (size_t)row * Kdim + ko,                        \
                         &Bs[BUF][row * QBLK + scol]);                           \
        }                                                                        \
    }
#define STAGE_COMMIT(BUF) wait_async0();
#else
#define STAGE_ISSUE(BUF, KB)                                                     \
    {                                                                            \
        _Pragma("unroll")                                                        \
        for (int r = 0; r < 2; ++r) {                                            \
            const int row = srow + 64 * r;                                       \
            const size_t ko = (size_t)(KB) * QBLK + scol;                        \
            ra[r] = *(const uint4*)(Abase + (size_t)row * Kdim + ko);            \
            rb[r] = *(const uint4*)(Bbase + (size_t)row * Kdim + ko);            \
        }                                                                        \
    }
#define STAGE_COMMIT(BUF)                                                        \
    {                                                                            \
        _Pragma("unroll")                                                        \
        for (int r = 0; r < 2; ++r) {                                            \
            const int row = srow + 64 * r;                                       \
            *(uint4*)&As[BUF][row * QBLK + scol] = ra[r];                        \
            *(uint4*)&Bs[BUF][row * QBLK + scol] = rb[r];                        \
        }                                                                        \
    }
#endif

#define COMPUTE_SLAB(BUF, KB)                                                    \
    {                                                                            \
        const float srowv = Ascale[(size_t)(m0 + wsm * 16 + llo) * kbc + (KB)];  \
        const float swv   = Bscale[(size_t)nb * kbc + (KB)];                     \
        float rs[8];                                                             \
        _Pragma("unroll")                                                        \
        for (int i = 0; i < 8; ++i)                                              \
            rs[i] = __shfl(srowv, i + (lhi << 3), 32) * swv;                     \
        v16i af;                                                                 \
        {                                                                        \
            const unsigned char* abase =                                         \
                &As[BUF][(wsm * 16 + llo) * QBLK + lhi * 16];                    \
            _Pragma("unroll")                                                    \
            for (int p = 0; p < 8; ++p) {                                        \
                const uint2 d = *(const uint2*)(abase + 32 * (p >> 1) + 8 * (p & 1)); \
                af[2 * p]     = (int)d.x;                                        \
                af[2 * p + 1] = (int)d.y;                                        \
            }                                                                    \
        }                                                                        \
        _Pragma("unroll")                                                        \
        for (int g = 0; g < 4; ++g) {                                            \
            v16i bf;                                                             \
            const unsigned char* bbase =                                         \
                &Bs[BUF][((wsn * 4 + g) * 16 + llo) * QBLK + lhi * 16];          \
            _Pragma("unroll")                                                    \
            for (int q = 0; q < 4; ++q) {                                        \
                const uint4 d = *(const uint4*)(bbase + 32 * q);                 \
                bf[4 * q]     = (int)d.x;                                        \
                bf[4 * q + 1] = (int)d.y;                                        \
                bf[4 * q + 2] = (int)d.z;                                        \
                bf[4 * q + 3] = (int)d.w;                                        \
            }                                                                    \
            v8f cz = {};                                                         \
            v8f t = __builtin_amdgcn_wmma_f32_16x16x128_fp8_fp8(                 \
                af, bf, (short)0, cz, false, false);                             \
            _Pragma("unroll")                                                    \
            for (int i = 0; i < 8; ++i)                                          \
                acc[g][i] += t[i] * rs[i];                                       \
        }                                                                        \
    }

    STAGE_ISSUE(0, 0)
    STAGE_COMMIT(0)
    __syncthreads();

    for (int kb = 0; kb < kbc; kb += 2) {
        const bool m1 = (kb + 1) < kbc;
        const bool m2 = (kb + 2) < kbc;
        if (m1) STAGE_ISSUE(1, kb + 1)
        COMPUTE_SLAB(0, kb)
        if (m1) STAGE_COMMIT(1)
        __syncthreads();
        if (m1) {
            if (m2) STAGE_ISSUE(0, kb + 2)
            COMPUTE_SLAB(1, kb + 1)
            if (m2) STAGE_COMMIT(0)
            __syncthreads();
        }
    }

#pragma unroll
    for (int g = 0; g < 4; ++g) {
        const int col = n0 + (wsn * 4 + g) * 16 + llo;
        const float bv = bias[col];
#pragma unroll
        for (int i = 0; i < 8; ++i) {
            const int row = m0 + wsm * 16 + i + (lhi << 3);
            out[(size_t)row * Ndim + col] = acc[g][i] + bv;
        }
    }
}

extern "C" void kernel_launch(void* const* d_in, const int* in_sizes, int n_in,
                              void* d_out, int out_size, void* d_ws, size_t ws_size,
                              hipStream_t stream) {
    (void)n_in; (void)out_size; (void)ws_size;
    const float* x    = (const float*)d_in[0];
    const float* w    = (const float*)d_in[1];
    const float* bias = (const float*)d_in[2];
    float* out = (float*)d_out;

    const int Nd  = in_sizes[2];
    const int Kd  = in_sizes[1] / Nd;
    const int Md  = in_sizes[0] / Kd;
    const int kbc = Kd / QBLK;
    const int NB  = Nd / QBLK;

    unsigned char* wsb = (unsigned char*)d_ws;
    unsigned char* xq  = wsb;
    unsigned char* wq  = xq + (size_t)Md * Kd;
    float* xs  = (float*)(wq + (size_t)Nd * Kd);
    float* wsc = xs + (size_t)Md * kbc;

    quant_act_kernel<<<(Md * kbc) / 8, 256, 0, stream>>>(x, xq, xs, Kd, kbc);
    quant_wgt_kernel<<<NB * kbc, 256, 0, stream>>>(w, wq, wsc, Kd, kbc);

    dim3 grid(Md / QBLK, Nd / QBLK);
    gemm_fp8_kernel<<<grid, 512, 0, stream>>>(xq, wq, xs, wsc, bias, out,
                                              Nd, Kd, kbc);
}
